// JTNNEncoder_2379411882633
// MI455X (gfx1250) — matrix operand order repaired and checked by reading, not compiled, not run on hardware
//
#include <hip/hip_runtime.h>
#include <hip/hip_bf16.h>
#include <stdint.h>

// Problem constants (match reference)
#define T_DEPTH 12
#define E_EDGES 8192
#define H_DIM   256
#define NB      8
#define B_ROOTS 256
#define M_MSGS  (T_DEPTH * E_EDGES)   // 98304

typedef __attribute__((ext_vector_type(16))) __bf16 v16bf;
typedef __attribute__((ext_vector_type(8)))  float  v8f;

union FragU { v16bf v; uint4 q[2]; };

__device__ __forceinline__ unsigned short f2bf(float f) {
  unsigned u = __float_as_uint(f);
  u += 0x7FFFu + ((u >> 16) & 1u);          // round-to-nearest-even
  return (unsigned short)(u >> 16);
}
__device__ __forceinline__ float bf2f(unsigned short s) {
  return __uint_as_float(((unsigned)s) << 16);
}
__device__ __forceinline__ unsigned pk2(float a, float b) {
  return (unsigned)f2bf(a) | ((unsigned)f2bf(b) << 16);
}
__device__ __forceinline__ float sigf(float x) { return 1.f / (1.f + __expf(-x)); }

// A-operand (16x32 bf16): lane m=lane&15, half=lane>>4 holds
// K = half*8+(0..7) in elems 0..7 and K = 16+half*8+(0..7) in elems 8..15.
__device__ __forceinline__ v16bf frag_a(const unsigned short* base, int stride, int k0) {
  int lane = threadIdx.x & 31;
  const unsigned short* p = base + (lane & 15) * stride + k0 + (lane >> 4) * 8;
  FragU f;
  f.q[0] = *(const uint4*)p;
  f.q[1] = *(const uint4*)(p + 16);
  return f.v;
}
// B-operand (32x16 bf16) loaded from pre-transposed weight WT[N][K]:
// lane n=lane&15 holds K = half*8+(0..7) in elems 0..7 and K = 16+half*8+(0..7) in elems 8..15.
__device__ __forceinline__ v16bf frag_b(const unsigned short* base, int stride, int k0) {
  int lane = threadIdx.x & 31;
  const unsigned short* p = base + (lane & 15) * stride + k0 + (lane >> 4) * 8;
  FragU f;
  f.q[0] = *(const uint4*)p;
  f.q[1] = *(const uint4*)(p + 16);
  return f.v;
}
#define WMMA_BF16(a, b, c) \
  __builtin_amdgcn_wmma_f32_16x16x32_bf16(false, (a), false, (b), (short)0, (c), false, false)

// ---------------- weight prep ----------------
// Generic: W[K][N] f32 row-major  ->  WT[N][K] bf16
__global__ void wt_convert(const float* __restrict__ W, unsigned short* __restrict__ WT,
                           int K, int N) {
  int idx = blockIdx.x * 256 + threadIdx.x;   // < N*K by grid construction
  int n = idx / K, k = idx - n * K;
  WT[idx] = f2bf(W[(size_t)k * N + n]);
}
// Fused [Wz | Wr] -> W1t[512][512] bf16 (Wr cols zero-padded for K>=256)
__global__ void w1_build(const float* __restrict__ Wz, const float* __restrict__ Wr,
                         unsigned short* __restrict__ W1t) {
  int idx = blockIdx.x * 256 + threadIdx.x;   // < 512*512
  int n = idx >> 9, k = idx & 511;
  float v;
  if (n < 256)      v = Wz[(size_t)k * 256 + n];
  else if (k < 256) v = Wr[(size_t)k * 256 + (n - 256)];
  else              v = 0.f;
  W1t[idx] = f2bf(v);
}
__global__ void zero_row0(float* __restrict__ h) { h[threadIdx.x] = 0.f; }

// ---------------- fused GRU step (one depth) ----------------
// block = 256 threads = 8 waves, owns 16 edges. grid = E/16 = 512.
// LDS layout (bytes):
//   xbf     @0       16x256 bf16   (8192)
//   sumbf   @8192    16x256 bf16   (8192)
//   hnbf    @16384  128x256 bf16  (65536)
//   gatedbf @81920   16x256 bf16   (8192)
//   sumf    @90112   16x256 f32   (16384)
//   r1f     @106496  16x256 f32   (16384)
//   zpf     @122880  16x256 f32   (16384)
#define SMEM_STEP 139264
__global__ __launch_bounds__(256) void gru_step(
    const int* __restrict__ x_ids, const int* __restrict__ nei_idx,
    const float* __restrict__ embedding,
    const unsigned short* __restrict__ W1t,   // [512][512]
    const unsigned short* __restrict__ Urt,   // [256][256]
    const unsigned short* __restrict__ Wht,   // [256][512]
    const float* __restrict__ bz, const float* __restrict__ bur,
    const float* __restrict__ bh,
    float* __restrict__ hglob, int t) {
  extern __shared__ unsigned char smem[];
  unsigned short* xbf     = (unsigned short*)(smem + 0);
  unsigned short* sumbf   = (unsigned short*)(smem + 8192);
  unsigned short* hnbf    = (unsigned short*)(smem + 16384);
  unsigned short* gatedbf = (unsigned short*)(smem + 81920);
  float*          sumf    = (float*)(smem + 90112);
  float*          r1f     = (float*)(smem + 106496);
  float*          zpf     = (float*)(smem + 122880);

  const int tid = threadIdx.x;
  const int lane = tid & 31, wave = tid >> 5;

  // ---- phase 1: gathers, f32->bf16, neighbor sum ----
  {
    int row = tid >> 4;            // local edge 0..15
    int col0 = (tid & 15) * 16;    // 16 columns per thread
    int edge = blockIdx.x * 16 + row;
    int xid = x_ids[(size_t)t * E_EDGES + edge];
    const float* xs = embedding + (size_t)xid * H_DIM + col0;
    for (int i = 0; i < 16; i += 4) {
      float4 v = *(const float4*)(xs + i);
      unsigned short* d = xbf + row * 256 + col0 + i;
      *(unsigned*)(d)     = pk2(v.x, v.y);
      *(unsigned*)(d + 2) = pk2(v.z, v.w);
    }
    float acc[16];
#pragma unroll
    for (int i = 0; i < 16; ++i) acc[i] = 0.f;
    const int* nrow = nei_idx + ((size_t)t * E_EDGES + edge) * NB;
    for (int nb = 0; nb < NB; ++nb) {
      int idx = nrow[nb];
      const float* hp = hglob + (size_t)idx * H_DIM + col0;
      unsigned short* d = hnbf + (row * NB + nb) * 256 + col0;
      for (int i = 0; i < 16; i += 4) {
        float4 v = *(const float4*)(hp + i);
        acc[i] += v.x; acc[i + 1] += v.y; acc[i + 2] += v.z; acc[i + 3] += v.w;
        *(unsigned*)(d + i)     = pk2(v.x, v.y);
        *(unsigned*)(d + i + 2) = pk2(v.z, v.w);
      }
    }
    for (int i = 0; i < 16; i += 4)
      *(float4*)(sumf + row * 256 + col0 + i) =
          make_float4(acc[i], acc[i + 1], acc[i + 2], acc[i + 3]);
    for (int i = 0; i < 16; i += 2)
      *(unsigned*)(sumbf + row * 256 + col0 + i) = pk2(acc[i], acc[i + 1]);
  }
  __syncthreads();

  // ---- phase 2: [x|sum_h] @ [Wz|Wr]  -> z_pre (n<256), r1 (n>=256) ----
  for (int j = 0; j < 4; ++j) {
    int nt = wave + j * 8;                       // 0..31
    v8f acc = {0.f, 0.f, 0.f, 0.f, 0.f, 0.f, 0.f, 0.f};
    const unsigned short* bbase = W1t + (size_t)nt * 16 * 512;
    for (int kt = 0; kt < 16; ++kt) {
      v16bf a = (kt < 8) ? frag_a(xbf, 256, kt * 32)
                         : frag_a(sumbf, 256, (kt - 8) * 32);
      v16bf b = frag_b(bbase, 512, kt * 32);
      acc = WMMA_BF16(a, b, acc);
    }
    int n = nt * 16 + (lane & 15);
    int half = lane >> 4;
    if (n < 256) {
#pragma unroll
      for (int r = 0; r < 8; ++r) zpf[(half * 8 + r) * 256 + n] = acc[r];
    } else {
#pragma unroll
      for (int r = 0; r < 8; ++r) r1f[(half * 8 + r) * 256 + (n - 256)] = acc[r];
    }
  }
  __syncthreads();

  // ---- phase 3: r2 = h_nei @ Ur ; r = sigmoid(r1+r2+bur); sum_gated ----
  {
    const unsigned short* abase = hnbf + wave * 16 * 256;  // 16 rows = edges 2w,2w+1
    for (int nt = 0; nt < 16; ++nt) {
      v8f acc = {0.f, 0.f, 0.f, 0.f, 0.f, 0.f, 0.f, 0.f};
      const unsigned short* bbase = Urt + (size_t)nt * 16 * 256;
      for (int kt = 0; kt < 8; ++kt) {
        v16bf a = frag_a(abase, 256, kt * 32);
        v16bf b = frag_b(bbase, 256, kt * 32);
        acc = WMMA_BF16(a, b, acc);
      }
      int n = nt * 16 + (lane & 15);
      int half = lane >> 4;
      int e = 2 * wave + half;                   // local edge for this lane's 8 rows
      float r1v = r1f[e * 256 + n];
      float burn = bur[n];
      float g = 0.f;
#pragma unroll
      for (int r = 0; r < 8; ++r) {
        float rv = sigf(r1v + burn + acc[r]);
        float hv = bf2f(hnbf[(wave * 16 + half * 8 + r) * 256 + n]);
        g += rv * hv;
      }
      gatedbf[e * 256 + n] = f2bf(g);
    }
  }
  __syncthreads();

  // ---- phase 4: pre_h = tanh([x|sum_gated]@Wh + bh); new_h out ----
  for (int j = 0; j < 2; ++j) {
    int nt = wave + j * 8;                       // 0..15
    v8f acc = {0.f, 0.f, 0.f, 0.f, 0.f, 0.f, 0.f, 0.f};
    const unsigned short* bbase = Wht + (size_t)nt * 16 * 512;
    for (int kt = 0; kt < 16; ++kt) {
      v16bf a = (kt < 8) ? frag_a(xbf, 256, kt * 32)
                         : frag_a(gatedbf, 256, (kt - 8) * 32);
      v16bf b = frag_b(bbase, 512, kt * 32);
      acc = WMMA_BF16(a, b, acc);
    }
    int n = nt * 16 + (lane & 15);
    int half = lane >> 4;
    float bhn = bh[n], bzn = bz[n];
#pragma unroll
    for (int r = 0; r < 8; ++r) {
      int e = half * 8 + r;
      float pre = tanhf(acc[r] + bhn);
      float zv = sigf(zpf[e * 256 + n] + bzn);
      float sh = sumf[e * 256 + n];
      hglob[(size_t)(1 + t * E_EDGES + blockIdx.x * 16 + e) * H_DIM + n] =
          (1.f - zv) * sh + zv * pre;
    }
  }
}

// ---------------- root aggregation ----------------
// 16 blocks x 16 roots; LDS: xbf 8KB + sumbf 8KB
#define SMEM_ROOT 16384
__global__ __launch_bounds__(256) void root_kernel(
    const int* __restrict__ root_wid, const int* __restrict__ root_nei,
    const float* __restrict__ embedding,
    const unsigned short* __restrict__ Wot,   // [256][512]
    const float* __restrict__ bo,
    const float* __restrict__ hglob, float* __restrict__ out) {
  extern __shared__ unsigned char smem[];
  unsigned short* xbf   = (unsigned short*)(smem + 0);
  unsigned short* sumbf = (unsigned short*)(smem + 8192);
  const int tid = threadIdx.x;
  const int lane = tid & 31, wave = tid >> 5;

  {
    int row = tid >> 4;
    int col0 = (tid & 15) * 16;
    int root = blockIdx.x * 16 + row;
    int wid = root_wid[root];
    const float* xs = embedding + (size_t)wid * H_DIM + col0;
    for (int i = 0; i < 16; i += 4) {
      float4 v = *(const float4*)(xs + i);
      unsigned short* d = xbf + row * 256 + col0 + i;
      *(unsigned*)(d)     = pk2(v.x, v.y);
      *(unsigned*)(d + 2) = pk2(v.z, v.w);
    }
    float acc[16];
#pragma unroll
    for (int i = 0; i < 16; ++i) acc[i] = 0.f;
    const int* nrow = root_nei + (size_t)root * NB;
    for (int nb = 0; nb < NB; ++nb) {
      int idx = nrow[nb];
      const float* hp = hglob + (size_t)idx * H_DIM + col0;
      for (int i = 0; i < 16; i += 4) {
        float4 v = *(const float4*)(hp + i);
        acc[i] += v.x; acc[i + 1] += v.y; acc[i + 2] += v.z; acc[i + 3] += v.w;
      }
    }
    for (int i = 0; i < 16; i += 2)
      *(unsigned*)(sumbf + row * 256 + col0 + i) = pk2(acc[i], acc[i + 1]);
  }
  __syncthreads();

  for (int j = 0; j < 2; ++j) {
    int nt = wave + j * 8;                       // 0..15
    v8f acc = {0.f, 0.f, 0.f, 0.f, 0.f, 0.f, 0.f, 0.f};
    const unsigned short* bbase = Wot + (size_t)nt * 16 * 512;
    for (int kt = 0; kt < 16; ++kt) {
      v16bf a = (kt < 8) ? frag_a(xbf, 256, kt * 32)
                         : frag_a(sumbf, 256, (kt - 8) * 32);
      v16bf b = frag_b(bbase, 512, kt * 32);
      acc = WMMA_BF16(a, b, acc);
    }
    int n = nt * 16 + (lane & 15);
    int half = lane >> 4;
    float bon = bo[n];
#pragma unroll
    for (int r = 0; r < 8; ++r) {
      int e = half * 8 + r;
      float v = acc[r] + bon;
      out[(size_t)(blockIdx.x * 16 + e) * H_DIM + n] = v > 0.f ? v : 0.f;
    }
  }
}

extern "C" void kernel_launch(void* const* d_in, const int* in_sizes, int n_in,
                              void* d_out, int out_size, void* d_ws, size_t ws_size,
                              hipStream_t stream) {
  const int*   x_ids     = (const int*)d_in[0];
  const int*   nei_idx   = (const int*)d_in[1];
  const int*   root_wid  = (const int*)d_in[2];
  const int*   root_nei  = (const int*)d_in[3];
  const float* embedding = (const float*)d_in[4];
  const float* Wz  = (const float*)d_in[5];
  const float* bz  = (const float*)d_in[6];
  const float* Wr  = (const float*)d_in[7];
  const float* Ur  = (const float*)d_in[8];
  const float* bur = (const float*)d_in[9];
  const float* Wh  = (const float*)d_in[10];
  const float* bh  = (const float*)d_in[11];
  const float* Wo  = (const float*)d_in[12];
  const float* bo  = (const float*)d_in[13];

  float* h     = (float*)d_out;                       // (1+M) x H, doubles as state
  float* rootv = h + (size_t)(1 + M_MSGS) * H_DIM;    // B x H

  unsigned short* W1t = (unsigned short*)d_ws;        // 512x512
  unsigned short* Urt = W1t + 512 * 512;              // 256x256
  unsigned short* Wht = Urt + 256 * 256;              // 256x512
  unsigned short* Wot = Wht + 256 * 512;              // 256x512

  (void)hipFuncSetAttribute((const void*)gru_step,
                            hipFuncAttributeMaxDynamicSharedMemorySize, SMEM_STEP);

  w1_build  <<<1024, 256, 0, stream>>>(Wz, Wr, W1t);
  wt_convert<<<256,  256, 0, stream>>>(Ur, Urt, 256, 256);
  wt_convert<<<512,  256, 0, stream>>>(Wh, Wht, 512, 256);
  wt_convert<<<512,  256, 0, stream>>>(Wo, Wot, 512, 256);
  zero_row0 <<<1,    256, 0, stream>>>(h);

  for (int t = 0; t < T_DEPTH; ++t)
    gru_step<<<E_EDGES / 16, 256, SMEM_STEP, stream>>>(
        x_ids, nei_idx, embedding, W1t, Urt, Wht, bz, bur, bh, h, t);

  root_kernel<<<B_ROOTS / 16, 256, SMEM_ROOT, stream>>>(
      root_wid, root_nei, embedding, Wot, bo, h, rootv);
}
